// PatchCrossAttentionFusion_33225867002489
// MI455X (gfx1250) — compile-verified, matrix operand order checked by reading, not run on hardware
//
#include <hip/hip_runtime.h>
#include <cmath>

typedef _Float16 half_t;
typedef __attribute__((ext_vector_type(16))) _Float16 v16h;
typedef __attribute__((ext_vector_type(8)))  _Float16 v8h;
typedef __attribute__((ext_vector_type(8)))  float    v8f;

constexpr int BIMG  = 4;
constexpr int C     = 384;
constexpr int Hdim  = 224;
constexpr int Wdim  = 224;
constexpr int PS    = 8;
constexpr int NHEAD = 12;
constexpr int HD    = 32;            // head dim
constexpr int NWH   = Hdim / PS;     // 28
constexpr int NTOK  = PS * PS;       // 64
constexpr int HW    = Hdim * Wdim;
constexpr float LN_EPS = 1e-5f;

// LDS layout (bytes). Regions are reused across phases.
constexpr int LNLF_OFF = 0;          // 64x384 f16 (48KB); later P heads 0..5
constexpr int LNHF_OFF = 49152;      // 64x384 f16 (48KB); later P heads 6..11
constexpr int P_OFF    = 0;          // per-head 64x64 f16 = 8KB, head*8192
constexpr int Q_OFF    = 98304;      // 64x384 f16 (48KB); later O; raw stage in ph.1
constexpr int K_OFF    = 147456;     // 64x384 f16 (48KB)
constexpr int VT_OFF   = 196608;     // 384x64 f16 transposed (48KB)
constexpr int SMEM_BYTES = 245760;   // 240KB
// Phase-1 staging: f32 [C][65] (pad to kill bank conflicts) = 99840B at Q_OFF
constexpr int RAW_PAD  = 65;

__device__ inline v16h load_frag(const half_t* base, int row, int rowstride,
                                 int kblock, int hh) {
  // A-fragment (16-bit, 16x32): lane&15 = row, halves 0..7 = K kblock+8*hh..,
  // halves 8..15 = K kblock+16+8*hh.. ; B-fragment mirrors with lane&15 = col.
  const half_t* p = base + row * rowstride + kblock + 8 * hh;
  union { v16h v; v8h h[2]; } u;
  u.h[0] = *(const v8h*)(p);
  u.h[1] = *(const v8h*)(p + 16);
  return u.v;
}

__device__ inline v8f wmma_f16(v16h a, v16h b, v8f c) {
  return __builtin_amdgcn_wmma_f32_16x16x32_f16(false, a, false, b, (short)0, c,
                                                false, false);
}

__global__ void convert_weights_kernel(const float* __restrict__ Wq,
                                       const float* __restrict__ Wkv,
                                       const float* __restrict__ Wo,
                                       half_t* __restrict__ dst) {
  const int NQ = C * C, NKV = 2 * C * C, NO = C * C;
  int i = blockIdx.x * blockDim.x + threadIdx.x;
  if (i < NQ)                 dst[i] = (half_t)Wq[i];
  else if (i < NQ + NKV)      dst[i] = (half_t)Wkv[i - NQ];
  else if (i < NQ + NKV + NO) dst[i] = (half_t)Wo[i - NQ - NKV];
}

__global__ void __launch_bounds__(384)
xattn_window_kernel(const float* __restrict__ y_lf, const float* __restrict__ y_hf,
                    const float* __restrict__ g_lf, const float* __restrict__ b_lf,
                    const float* __restrict__ g_hf, const float* __restrict__ b_hf,
                    const half_t* __restrict__ Wq16, const float* __restrict__ bq,
                    const half_t* __restrict__ Wkv16, const float* __restrict__ bkv,
                    const half_t* __restrict__ Wo16, const float* __restrict__ bo,
                    float* __restrict__ out) {
  extern __shared__ char smem_raw[];

  const int wg   = blockIdx.x;
  const int bimg = wg / (NWH * NWH);
  const int rem  = wg % (NWH * NWH);
  const int h0   = (rem / NWH) * PS;
  const int w0   = (rem % NWH) * PS;

  const int tid  = threadIdx.x;
  const int wave = tid >> 5;
  const int lane = tid & 31;
  const int l15  = lane & 15;
  const int hh   = lane >> 4;

  const size_t imgbase = (size_t)bimg * C * HW + (size_t)h0 * Wdim + w0;

  half_t* lnLF = (half_t*)(smem_raw + LNLF_OFF);
  half_t* lnHF = (half_t*)(smem_raw + LNHF_OFF);
  half_t* Qb   = (half_t*)(smem_raw + Q_OFF);
  half_t* Kb   = (half_t*)(smem_raw + K_OFF);
  half_t* Vt   = (half_t*)(smem_raw + VT_OFF);
  half_t* Ob   = Qb;                       // reuse Q region after scores done
  float*  raw  = (float*)(smem_raw + Q_OFF); // phase-1 staging [C][RAW_PAD]

  // ------- Phase 1: coalesced stage -> LDS, then LayerNorm lf & hf (f16) ----
  for (int pass = 0; pass < 2; ++pass) {
    const float* src = pass ? y_hf : y_lf;
    const float* gg  = pass ? g_hf : g_lf;
    const float* bb  = pass ? b_hf : b_lf;
    half_t* dst      = pass ? lnHF : lnLF;

    // stage raw window: consecutive tids read contiguous 32B (tx fastest)
    for (int e = tid; e < C * NTOK; e += 384) {
      const int c = e >> 6;            // channel
      const int t = e & 63;            // token
      const int ty = t >> 3, tx = t & 7;
      raw[c * RAW_PAD + t] = src[imgbase + (size_t)c * HW + (size_t)ty * Wdim + tx];
    }
    __syncthreads();

    // LN per token: one wave per token, conflict-free strided LDS reads
    for (int t = wave; t < NTOK; t += 12) {
      float v[12];
      float s = 0.f, s2 = 0.f;
#pragma unroll
      for (int i = 0; i < 12; ++i) {
        const int c = lane + 32 * i;
        const float x = raw[c * RAW_PAD + t];
        v[i] = x; s += x; s2 += x * x;
      }
#pragma unroll
      for (int m = 16; m >= 1; m >>= 1) {
        s  += __shfl_xor(s,  m, 32);
        s2 += __shfl_xor(s2, m, 32);
      }
      const float mean = s * (1.f / C);
      const float var  = s2 * (1.f / C) - mean * mean;
      const float rstd = rsqrtf(var + LN_EPS);
#pragma unroll
      for (int i = 0; i < 12; ++i) {
        const int c = lane + 32 * i;
        dst[t * C + c] = (half_t)((v[i] - mean) * rstd * gg[c] + bb[c]);
      }
    }
    __syncthreads();   // raw region about to be reused (hf stage / Qb writes)
  }

  // ---------------- Phase 2: Q = LN(lf) @ Wq^T + bq  (to Qb, f16) -----------
  for (int tidx = wave; tidx < 4 * 24; tidx += 12) {
    const int mt = tidx / 24, nt = tidx % 24;
    v8f acc = {};
#pragma unroll
    for (int kb = 0; kb < 12; ++kb) {
      v16h a   = load_frag(lnLF, 16 * mt + l15, C, 32 * kb, hh);
      v16h bfr = load_frag(Wq16, 16 * nt + l15, C, 32 * kb, hh);
      acc = wmma_f16(a, bfr, acc);
    }
    const int n = 16 * nt + l15;
    const float bias = bq[n];
#pragma unroll
    for (int r = 0; r < 8; ++r) {
      const int m = 16 * mt + 8 * hh + r;
      Qb[m * C + n] = (half_t)(acc[r] + bias);
    }
  }

  // ---------------- Phase 3: KV = LN(hf) @ Wkv^T + bkv (K row-major, V^T) ---
  for (int tidx = wave; tidx < 4 * 48; tidx += 12) {
    const int mt = tidx / 48, nt = tidx % 48;
    v8f acc = {};
#pragma unroll
    for (int kb = 0; kb < 12; ++kb) {
      v16h a   = load_frag(lnHF, 16 * mt + l15, C, 32 * kb, hh);
      v16h bfr = load_frag(Wkv16, 16 * nt + l15, C, 32 * kb, hh);
      acc = wmma_f16(a, bfr, acc);
    }
    const int j = 16 * nt + l15;
    const float bias = bkv[j];
    if (j < C) {                       // K channels: row-major [token][384]
#pragma unroll
      for (int r = 0; r < 8; ++r) {
        const int m = 16 * mt + 8 * hh + r;
        Kb[m * C + j] = (half_t)(acc[r] + bias);
      }
    } else {                           // V channels: transposed [384][token]
      const int jc = j - C;
      v8h pk;
#pragma unroll
      for (int r = 0; r < 8; ++r) pk[r] = (half_t)(acc[r] + bias);
      *(v8h*)(&Vt[jc * NTOK + 16 * mt + 8 * hh]) = pk;   // 8 contiguous tokens
    }
  }
  __syncthreads();

  // ---------------- Phase 4: per-head attention (wave == head) --------------
  const int head = wave;
  const float scale = 0.17677669529663687f;  // 1/sqrt(32)

  // scores S = Q_h K_h^T : 4x4 tiles, K=32 (= head dim) in one WMMA each
  v8f sc[4][4];
#pragma unroll
  for (int i = 0; i < 4; ++i) {
    v16h a = load_frag(Qb, 16 * i + l15, C, head * HD, hh);
#pragma unroll
    for (int j = 0; j < 4; ++j) {
      v16h bfr = load_frag(Kb, 16 * j + l15, C, head * HD, hh);
      v8f z = {};
      sc[i][j] = wmma_f16(a, bfr, z);
    }
  }

  // softmax across rows (row = vgpr+halfgroup; reduce within 16-lane groups)
  half_t* Pb = (half_t*)(smem_raw + P_OFF) + head * (NTOK * NTOK);
#pragma unroll
  for (int i = 0; i < 4; ++i) {
#pragma unroll
    for (int r = 0; r < 8; ++r) {
      float mx = fmaxf(fmaxf(sc[i][0][r], sc[i][1][r]),
                       fmaxf(sc[i][2][r], sc[i][3][r]));
#pragma unroll
      for (int m = 8; m >= 1; m >>= 1) mx = fmaxf(mx, __shfl_xor(mx, m, 32));
      float sum = 0.f;
#pragma unroll
      for (int j = 0; j < 4; ++j) {
        const float e = __expf((sc[i][j][r] - mx) * scale);
        sc[i][j][r] = e; sum += e;
      }
#pragma unroll
      for (int m = 8; m >= 1; m >>= 1) sum += __shfl_xor(sum, m, 32);
      const float inv = 1.f / sum;
      const int mrow = 16 * i + 8 * hh + r;
#pragma unroll
      for (int j = 0; j < 4; ++j)
        Pb[mrow * NTOK + 16 * j + l15] = (half_t)(sc[i][j][r] * inv);
    }
  }
  __syncthreads();   // all Q reads done; P fully staged

  // out_h = P @ V_h : 4x2 tiles, K-loop of 2 (B-fragments contiguous via Vt)
#pragma unroll
  for (int i = 0; i < 4; ++i) {
#pragma unroll
    for (int d = 0; d < 2; ++d) {
      v8f acc = {};
#pragma unroll
      for (int kb = 0; kb < 2; ++kb) {
        v16h a   = load_frag(Pb, 16 * i + l15, NTOK, 32 * kb, hh);
        v16h bfr = load_frag(Vt, head * HD + 16 * d + l15, NTOK, 32 * kb, hh);
        acc = wmma_f16(a, bfr, acc);
      }
      const int n = head * HD + 16 * d + l15;
#pragma unroll
      for (int r = 0; r < 8; ++r) {
        const int m = 16 * i + 8 * hh + r;
        Ob[m * C + n] = (half_t)acc[r];    // Ob aliases Qb (safe after barrier)
      }
    }
  }
  __syncthreads();

  // ---------------- Phase 5: Y = O @ Wo^T + bo + hf ; window_reverse store --
  for (int tidx = wave; tidx < 4 * 24; tidx += 12) {
    const int mt = tidx / 24, nt = tidx % 24;
    v8f acc = {};
#pragma unroll
    for (int kb = 0; kb < 12; ++kb) {
      v16h a   = load_frag(Ob, 16 * mt + l15, C, 32 * kb, hh);
      v16h bfr = load_frag(Wo16, 16 * nt + l15, C, 32 * kb, hh);
      acc = wmma_f16(a, bfr, acc);
    }
    const int c = 16 * nt + l15;
    const float bias = bo[c];
#pragma unroll
    for (int r = 0; r < 8; ++r) {
      const int m  = 16 * mt + 8 * hh + r;
      const int ty = m >> 3, tx = m & 7;
      const size_t gidx = imgbase + (size_t)c * HW + (size_t)ty * Wdim + tx;
      out[gidx] = acc[r] + bias + y_hf[gidx];   // residual from HF; contiguous
    }
  }
}

extern "C" void kernel_launch(void* const* d_in, const int* in_sizes, int n_in,
                              void* d_out, int out_size, void* d_ws, size_t ws_size,
                              hipStream_t stream) {
  const float* y_lf = (const float*)d_in[0];
  const float* y_hf = (const float*)d_in[1];
  const float* g_lf = (const float*)d_in[2];
  const float* b_lf = (const float*)d_in[3];
  const float* g_hf = (const float*)d_in[4];
  const float* b_hf = (const float*)d_in[5];
  const float* Wq   = (const float*)d_in[6];
  const float* bq   = (const float*)d_in[7];
  const float* Wkv  = (const float*)d_in[8];
  const float* bkv  = (const float*)d_in[9];
  const float* Wo   = (const float*)d_in[10];
  const float* bo   = (const float*)d_in[11];
  float* out = (float*)d_out;

  half_t* w16 = (half_t*)d_ws;                 // [Wq16 | Wkv16 | Wo16], 1.15MB
  const int NW = C * C + 2 * C * C + C * C;    // 589824
  convert_weights_kernel<<<(NW + 255) / 256, 256, 0, stream>>>(Wq, Wkv, Wo, w16);

  (void)hipFuncSetAttribute((const void*)xattn_window_kernel,
                            hipFuncAttributeMaxDynamicSharedMemorySize, SMEM_BYTES);

  const int nwin = BIMG * NWH * NWH;           // 3136 windows
  xattn_window_kernel<<<nwin, 384, SMEM_BYTES, stream>>>(
      y_lf, y_hf, g_lf, b_lf, g_hf, b_hf,
      w16, bq, w16 + C * C, bkv, w16 + 3 * C * C, bo, out);
}
